// ExtendedModel2_58695023067739
// MI455X (gfx1250) — matrix operand order repaired and checked by reading, not compiled, not run on hardware
//
#include <hip/hip_runtime.h>
#include <hip/hip_bf16.h>

typedef __attribute__((ext_vector_type(16))) _Float16 v16h;
typedef __attribute__((ext_vector_type(8)))  float    v8f;
typedef __attribute__((ext_vector_type(4)))  unsigned v4u;
typedef __attribute__((ext_vector_type(8)))  int      v8i;
typedef __attribute__((ext_vector_type(4)))  int      v4i;

#if defined(__has_builtin)
#  if __has_builtin(__builtin_amdgcn_tensor_load_to_lds)
#    define HAVE_TDM 1
#  else
#    define HAVE_TDM 0
#  endif
#else
#  define HAVE_TDM 0
#endif

struct __align__(16) H8 { _Float16 h[8]; };

// ---------------------------------------------------------------------------
// Fragment loaders: contiguous k order, element s of lane l holds k = 16*(l/16) + s (A and B)
// ---------------------------------------------------------------------------
__device__ __forceinline__ v16h frag_a_f16(const _Float16* arow, int kb8) {
    H8 a0 = *(const H8*)(arow + 2 * kb8);
    H8 a1 = *(const H8*)(arow + 2 * kb8 + 8);
    v16h a;
#pragma unroll
    for (int i = 0; i < 8; ++i) { a[i] = a0.h[i]; a[8 + i] = a1.h[i]; }
    return a;
}

__device__ __forceinline__ v16h frag_a_f32(const float* arow, int kb8) {
    float4 x0 = *(const float4*)(arow + 2 * kb8);
    float4 x1 = *(const float4*)(arow + 2 * kb8 + 4);
    float4 y0 = *(const float4*)(arow + 2 * kb8 + 8);
    float4 y1 = *(const float4*)(arow + 2 * kb8 + 12);
    float t[16];
    *(float4*)(t + 0) = x0; *(float4*)(t + 4)  = x1;
    *(float4*)(t + 8) = y0; *(float4*)(t + 12) = y1;
    v16h a;
#pragma unroll
    for (int i = 0; i < 16; ++i) a[i] = (_Float16)t[i];
    return a;
}

__device__ __forceinline__ v16h frag_b(const _Float16* bcol) {
    H8 b0 = *(const H8*)(bcol);
    H8 b1 = *(const H8*)(bcol + 8);
    v16h b;
#pragma unroll
    for (int i = 0; i < 8; ++i) { b[i] = b0.h[i]; b[8 + i] = b1.h[i]; }
    return b;
}

// ---------------------------------------------------------------------------
// TDM 2-D tile load: global (tile_y rows x tile_x f16, row stride ldb elems)
// -> LDS at lds_off with hardware row padding 32->40 halves (pad_interval=16
// DWORDs, pad_amount=4 DWORDs), matching BS_STRIDE bank-conflict layout.
// D# per CDNA5 ISA (group0 + group1, 2-D => groups 2/3
// zero). 6-arg builtin form (this toolchain): extra zero int32x8 operand.
// ---------------------------------------------------------------------------
#if HAVE_TDM
__device__ __forceinline__ void tdm_load_2d(unsigned lds_off, const void* gaddr,
                                            unsigned tile_x, unsigned tile_y,
                                            unsigned stride_elems) {
    unsigned long long ga = (unsigned long long)gaddr;
    v4u g0;
    g0[0] = 1u;                                  // count=1 (valid descriptor)
    g0[1] = lds_off;                             // lds_addr (bytes)
    g0[2] = (unsigned)ga;                        // global_addr[31:0]
    g0[3] = (unsigned)((ga >> 32) & 0x1FFFFFFu) | (2u << 30);  // addr[56:32] | type=2
    v8i g1;
    g1[0] = (int)((1u << 16) | (1u << 20) | (3u << 22) | (3u << 25));
    //            data_size=2B  pad_enable   pad_int=16DW  pad_amt=4DW
    g1[1] = (int)((stride_elems & 0xFFFFu) << 16);   // tensor_dim0 lo16 (OOB-free)
    g1[2] = (int)((stride_elems >> 16) | (0xFFFFu << 16)); // dim0 hi | tensor_dim1 lo16
    g1[3] = (int)(tile_x << 16);                 // tensor_dim1 hi=0 | tile_dim0
    g1[4] = (int)tile_y;                         // tile_dim1 | tile_dim2=0
    g1[5] = (int)stride_elems;                   // tensor_dim0_stride lo32
    g1[6] = 0;                                   // stride hi | dim1_stride lo (unused, 2-D)
    g1[7] = 0;
    v4i z4 = {};
    v8i z8 = {};
    __builtin_amdgcn_tensor_load_to_lds(g0, g1, z4, z4, z8, 0);
}
#endif

// ---------------------------------------------------------------------------
// Kernel 1: max-pool over (B*512) rows of 2048 f32 -> gfeat f16 (32x512)
// ---------------------------------------------------------------------------
__global__ __launch_bounds__(256) void maxpool_kernel(const float* __restrict__ gf,
                                                      _Float16* __restrict__ gfeat,
                                                      int rows, int mlen) {
    int wave = (blockIdx.x * blockDim.x + threadIdx.x) >> 5;
    int lane = threadIdx.x & 31;
    if (wave >= rows) return;
    const float4* p = reinterpret_cast<const float4*>(gf + (long long)wave * mlen);
    int nvec = mlen >> 2;
    float m = -3.402823466e38f;
    for (int i = lane; i < nvec; i += 32) {
        float4 v = p[i];
        m = fmaxf(m, fmaxf(fmaxf(v.x, v.y), fmaxf(v.z, v.w)));
    }
#pragma unroll
    for (int off = 16; off; off >>= 1) m = fmaxf(m, __shfl_xor(m, off, 32));
    if (lane == 0) gfeat[wave] = (_Float16)m;
}

// ---------------------------------------------------------------------------
// Kernel 2: token BN scale/shift
// ---------------------------------------------------------------------------
__global__ void prep_tok_bn(const float* __restrict__ g, const float* __restrict__ b,
                            const float* __restrict__ m, const float* __restrict__ v,
                            float* __restrict__ s, float* __restrict__ t, int n) {
    int i = blockIdx.x * blockDim.x + threadIdx.x;
    if (i < n) {
        float sc = g[i] * rsqrtf(v[i] + 1e-5f);
        s[i] = sc;
        t[i] = b[i] - m[i] * sc;
    }
}

// ---------------------------------------------------------------------------
// Kernel 3: fold BN into weights, pad to WMMA tiles, convert f16.
// ---------------------------------------------------------------------------
__global__ void prep_w(const float* __restrict__ W, const float* __restrict__ bias_in,
                       const float* __restrict__ g, const float* __restrict__ bb,
                       const float* __restrict__ mm, const float* __restrict__ vv,
                       _Float16* __restrict__ Wo, float* __restrict__ Bo,
                       int Mreal, int Mpad, int Kreal, int Kpad) {
    int total = Mpad * Kpad;
    int stride = gridDim.x * blockDim.x;
    for (int i = blockIdx.x * blockDim.x + threadIdx.x; i < total + Mpad; i += stride) {
        if (i < total) {
            int r = i / Kpad, k = i % Kpad;
            float w = 0.f;
            if (r < Mreal && k < Kreal) {
                w = W[r * Kreal + k];
                if (g) w *= g[r] * rsqrtf(vv[r] + 1e-5f);
            }
            Wo[i] = (_Float16)w;
        } else {
            int r = i - total;
            float bo = 0.f;
            if (r < Mreal) {
                if (g) { float sc = g[r] * rsqrtf(vv[r] + 1e-5f); bo = bb[r] - mm[r] * sc; }
                else bo = bias_in[r];
            }
            Bo[r] = bo;
        }
    }
}

// ---------------------------------------------------------------------------
// Kernel 4: token GEMM (M=65536, N=32, K=512) + fused BN/ReLU epilogue.
// ---------------------------------------------------------------------------
__global__ __launch_bounds__(256) void token_kernel(const float* __restrict__ token_w,
                                                    const float* __restrict__ token_b,
                                                    const float* __restrict__ s_tok,
                                                    const float* __restrict__ t_tok,
                                                    const _Float16* __restrict__ gfeat,
                                                    _Float16* __restrict__ tokT) {
    int lane = threadIdx.x & 31;
    int wave = threadIdx.x >> 5;
    int tile = blockIdx.x * 8 + wave;
    int mt = tile >> 1, nt = tile & 1;
    int m = lane & 15;
    int kb8  = (lane >> 4) << 3;
    int kb16 = (lane >> 4) << 4;
    int n = lane & 15;
    const float*    arow = token_w + ((long long)mt * 16 + m) * 512;
    const _Float16* bcol = gfeat + ((long long)(nt * 16 + n)) * 512;
    v8f acc = {};
#pragma unroll 4
    for (int k0 = 0; k0 < 512; k0 += 32) {
        __builtin_prefetch(arow + k0 + 256, 0, 1);   // global_prefetch_b8
        v16h a = frag_a_f32(arow + k0, kb8);
        v16h b = frag_b(bcol + k0 + kb16);
        acc = __builtin_amdgcn_wmma_f32_16x16x32_f16(false, a, false, b, (short)0, acc,
                                                     false, false);
    }
    int bcolidx = nt * 16 + (lane & 15);
#pragma unroll
    for (int r = 0; r < 8; ++r) {
        int j = mt * 16 + r + ((lane >> 4) << 3);
        float y = fmaf(acc[r] + token_b[j], s_tok[j], t_tok[j]);
        y = fmaxf(y, 0.f);
        int c = j >> 10, nn = j & 1023;
        tokT[(((long long)bcolidx << 10) + nn) * 64 + c] = (_Float16)y;
    }
}

#define BS_STRIDE 40

// ---------------------------------------------------------------------------
// Kernel 5: generic batched WMMA GEMM, double-buffered async global->LDS B
// staging (ASYNCcnt pipelined: issue chunk i+1, wait <=1 so chunk i landed).
// Block = 8 waves covering 8mt x 4nt; each wave: 1 A frag -> 4 WMMAs.
// ---------------------------------------------------------------------------
__global__ __launch_bounds__(256) void wmma_gemm_kernel(
    const _Float16* __restrict__ A, int lda, long long sA,
    const _Float16* __restrict__ Bt, int ldb, long long sB,
    void* __restrict__ Dv, int ldd, long long sD,
    const float* __restrict__ bias,
    int tilesM, int tilesN, int K, int Mreal, int act, int flags) {
    __shared__ _Float16 Bs[2][64 * BS_STRIDE];
    int lane = threadIdx.x & 31;
    int wave = threadIdx.x >> 5;
    int blocksM = (tilesM + 7) >> 3;
    int bm = blockIdx.x % blocksM;
    int bn = blockIdx.x / blocksM;
    bool active = (bm * 8 + wave) < tilesM;
    int mt = bm * 8 + wave;
    if (!active) mt = tilesM - 1;            // clamped redundant compute, store masked
    int ntbase = bn * 4;
    int batch = blockIdx.y;
    int m = lane & 15;
    int kb8  = (lane >> 4) << 3;
    int kb16 = (lane >> 4) << 4;
    const _Float16* arow = A + batch * sA + ((long long)(mt * 16 + m)) * lda;
    int sn = threadIdx.x >> 2;
    int sp = (threadIdx.x & 3) << 3;
    const _Float16* gsrc = Bt + batch * sB + ((long long)(ntbase * 16 + sn)) * ldb + sp;
    _Float16* ldst0 = &Bs[0][sn * BS_STRIDE + sp];
    _Float16* ldst1 = &Bs[1][sn * BS_STRIDE + sp];
    v8f acc[4] = {};
    int nch = K >> 5;
    // prologue: fill both buffers
    asm volatile("global_load_async_to_lds_b128 %0, %1, off"
                 :: "v"((unsigned)(unsigned long long)ldst0), "v"(gsrc) : "memory");
    if (nch > 1)
        asm volatile("global_load_async_to_lds_b128 %0, %1, off"
                     :: "v"((unsigned)(unsigned long long)ldst1), "v"(gsrc + 32) : "memory");
    for (int i = 0; i < nch; ++i) {
        if (i + 1 < nch) asm volatile("s_wait_asynccnt 0x1" ::: "memory");
        else             asm volatile("s_wait_asynccnt 0x0" ::: "memory");
        __syncthreads();
        v16h a = frag_a_f16(arow + (i << 5), kb8);
        const _Float16* bufp = &Bs[i & 1][0];
#pragma unroll
        for (int j = 0; j < 4; ++j) {
            v16h b = frag_b(bufp + (j * 16 + (lane & 15)) * BS_STRIDE + kb16);
            acc[j] = __builtin_amdgcn_wmma_f32_16x16x32_f16(false, a, false, b, (short)0,
                                                            acc[j], false, false);
        }
        __syncthreads();
        if (i + 2 < nch) {
            _Float16* ldst = (i & 1) ? ldst1 : ldst0;
            asm volatile("global_load_async_to_lds_b128 %0, %1, off"
                         :: "v"((unsigned)(unsigned long long)ldst),
                            "v"(gsrc + ((i + 2) << 5)) : "memory");
        }
    }
    if (!active) return;
    bool storeT = flags & 1, f32out = flags & 2;
#pragma unroll
    for (int j = 0; j < 4; ++j) {
        int col = (ntbase + j) * 16 + (lane & 15);
#pragma unroll
        for (int r = 0; r < 8; ++r) {
            int row = mt * 16 + r + ((lane >> 4) << 3);
            if (row >= Mreal) continue;
            float y = acc[j][r];
            if (bias) y += bias[row];
            if (act == 1) y = fmaxf(y, 0.f);
            else if (act == 2) y = 1.f / (1.f + __expf(-y)) - 0.5f;
            long long idx = (long long)batch * sD +
                            (storeT ? (long long)col * ldd + row : (long long)row * ldd + col);
            if (f32out) ((float*)Dv)[idx] = y;
            else        ((_Float16*)Dv)[idx] = (_Float16)y;
        }
    }
}

// ---------------------------------------------------------------------------
// Kernel 6: KV = k @ v^T (M=128, N=128, K=1024) per batch, TDM-staged B.
// Wave 0 issues one tensor_load_to_lds per 64x32 chunk (double-buffered,
// TENSORcnt pipelined); all waves consume from LDS after the barrier.
// Output stored transposed: KVt[c][c'].
// ---------------------------------------------------------------------------
__global__ __launch_bounds__(256) void kv_kernel(const _Float16* __restrict__ kC,
                                                 const _Float16* __restrict__ vC,
                                                 _Float16* __restrict__ KVt) {
    __shared__ _Float16 Bs[2][64 * BS_STRIDE];
    int lane = threadIdx.x & 31;
    int wave = threadIdx.x >> 5;
    int batch = blockIdx.y;
    int ntbase = blockIdx.x * 4;             // 2 blocks cover 8 n-tiles
    int mt = wave;                           // 8 m-tiles
    int m = lane & 15;
    int kb8  = (lane >> 4) << 3;
    int kb16 = (lane >> 4) << 4;
    const _Float16* arow = kC + (long long)batch * 131072 + ((long long)(mt * 16 + m)) * 1024;
    const _Float16* bbase = vC + (long long)batch * 131072 + ((long long)(ntbase * 16)) * 1024;
#if !HAVE_TDM
    int sn = threadIdx.x >> 2;
    int sp = (threadIdx.x & 3) << 3;
#endif
    v8f acc[4] = {};
    const int nch = 32;
#if HAVE_TDM
    if (wave == 0) {
        tdm_load_2d((unsigned)(unsigned long long)&Bs[0][0], bbase, 32, 64, 1024);
        tdm_load_2d((unsigned)(unsigned long long)&Bs[1][0], bbase + 32, 32, 64, 1024);
    }
#else
    asm volatile("global_load_async_to_lds_b128 %0, %1, off"
                 :: "v"((unsigned)(unsigned long long)&Bs[0][sn * BS_STRIDE + sp]),
                    "v"(bbase + (long long)sn * 1024 + sp) : "memory");
    asm volatile("global_load_async_to_lds_b128 %0, %1, off"
                 :: "v"((unsigned)(unsigned long long)&Bs[1][sn * BS_STRIDE + sp]),
                    "v"(bbase + (long long)sn * 1024 + sp + 32) : "memory");
#endif
    for (int i = 0; i < nch; ++i) {
#if HAVE_TDM
        if (wave == 0) {
            if (i + 1 < nch) __builtin_amdgcn_s_wait_tensorcnt(1);
            else             __builtin_amdgcn_s_wait_tensorcnt(0);
        }
#else
        if (i + 1 < nch) asm volatile("s_wait_asynccnt 0x1" ::: "memory");
        else             asm volatile("s_wait_asynccnt 0x0" ::: "memory");
#endif
        __syncthreads();
        v16h a = frag_a_f16(arow + (i << 5), kb8);
        const _Float16* bufp = &Bs[i & 1][0];
#pragma unroll
        for (int j = 0; j < 4; ++j) {
            v16h b = frag_b(bufp + (j * 16 + (lane & 15)) * BS_STRIDE + kb16);
            acc[j] = __builtin_amdgcn_wmma_f32_16x16x32_f16(false, a, false, b, (short)0,
                                                            acc[j], false, false);
        }
        __syncthreads();
        if (i + 2 < nch) {
#if HAVE_TDM
            if (wave == 0)
                tdm_load_2d((unsigned)(unsigned long long)&Bs[i & 1][0],
                            bbase + ((i + 2) << 5), 32, 64, 1024);
#else
            asm volatile("global_load_async_to_lds_b128 %0, %1, off"
                         :: "v"((unsigned)(unsigned long long)&Bs[i & 1][sn * BS_STRIDE + sp]),
                            "v"(bbase + (long long)sn * 1024 + sp + ((i + 2) << 5)) : "memory");
#endif
        }
    }
#pragma unroll
    for (int j = 0; j < 4; ++j) {
        int col = (ntbase + j) * 16 + (lane & 15);
#pragma unroll
        for (int r = 0; r < 8; ++r) {
            int row = mt * 16 + r + ((lane >> 4) << 3);
            KVt[(long long)batch * 16384 + (long long)col * 128 + row] = (_Float16)acc[j][r];
        }
    }
}

// ---------------------------------------------------------------------------
extern "C" void kernel_launch(void* const* d_in, const int* in_sizes, int n_in,
                              void* d_out, int out_size, void* d_ws, size_t ws_size,
                              hipStream_t stream) {
    (void)in_sizes; (void)n_in; (void)out_size; (void)ws_size;
    const float* global_f = (const float*)d_in[0];
    const float* token_w = (const float*)d_in[3];
    const float* token_b = (const float*)d_in[4];
    const float* tbn_g = (const float*)d_in[5], *tbn_b = (const float*)d_in[6];
    const float* tbn_m = (const float*)d_in[7], *tbn_v = (const float*)d_in[8];
    const float* q_w = (const float*)d_in[9];
    const float* qbn_g = (const float*)d_in[10], *qbn_b = (const float*)d_in[11];
    const float* qbn_m = (const float*)d_in[12], *qbn_v = (const float*)d_in[13];
    const float* k_w = (const float*)d_in[14];
    const float* kbn_g = (const float*)d_in[15], *kbn_b = (const float*)d_in[16];
    const float* kbn_m = (const float*)d_in[17], *kbn_v = (const float*)d_in[18];
    const float* v_w = (const float*)d_in[19];
    const float* vbn_g = (const float*)d_in[20], *vbn_b = (const float*)d_in[21];
    const float* vbn_m = (const float*)d_in[22], *vbn_v = (const float*)d_in[23];
    const float* f_w = (const float*)d_in[24];
    const float* fbn_g = (const float*)d_in[25], *fbn_b = (const float*)d_in[26];
    const float* fbn_m = (const float*)d_in[27], *fbn_v = (const float*)d_in[28];
    const float* c1_w = (const float*)d_in[29], *c1_b = (const float*)d_in[30];
    const float* c2_w = (const float*)d_in[31], *c2_b = (const float*)d_in[32];
    const float* c3_w = (const float*)d_in[33], *c3_b = (const float*)d_in[34];

    char* ws = (char*)d_ws;
    size_t off = 0;
    auto alloc = [&](size_t bytes) -> char* {
        char* p = ws + off;
        off += bytes;
        off = (off + 255) & ~(size_t)255;
        return p;
    };
    _Float16* gfeat = (_Float16*)alloc((size_t)32 * 512 * 2);
    float* s_tok = (float*)alloc((size_t)65536 * 4);
    float* t_tok = (float*)alloc((size_t)65536 * 4);
    _Float16* tokT = (_Float16*)alloc((size_t)32 * 1024 * 64 * 2);
    _Float16* Wq = (_Float16*)alloc(128 * 64 * 2); float* Bq = (float*)alloc(128 * 4);
    _Float16* Wk = (_Float16*)alloc(128 * 64 * 2); float* Bk = (float*)alloc(128 * 4);
    _Float16* Wv = (_Float16*)alloc(128 * 64 * 2); float* Bv = (float*)alloc(128 * 4);
    _Float16* Wf = (_Float16*)alloc(64 * 128 * 2); float* Bf = (float*)alloc(64 * 4);
    _Float16* W1 = (_Float16*)alloc(256 * 64 * 2); float* B1 = (float*)alloc(256 * 4);
    _Float16* W2 = (_Float16*)alloc(64 * 256 * 2); float* B2 = (float*)alloc(64 * 4);
    _Float16* W3 = (_Float16*)alloc(16 * 64 * 2);  float* B3 = (float*)alloc(16 * 4);
    _Float16* qT   = (_Float16*)alloc((size_t)32 * 1024 * 128 * 2);
    _Float16* kC   = (_Float16*)alloc((size_t)32 * 128 * 1024 * 2);
    _Float16* vC   = (_Float16*)alloc((size_t)32 * 128 * 1024 * 2);
    _Float16* KVt  = (_Float16*)alloc((size_t)32 * 128 * 128 * 2);
    _Float16* valT = (_Float16*)alloc((size_t)32 * 1024 * 128 * 2);
    _Float16* fusT = (_Float16*)alloc((size_t)32 * 1024 * 64 * 2);
    _Float16* h1T  = (_Float16*)alloc((size_t)32 * 1024 * 256 * 2);
    _Float16* h2T  = (_Float16*)alloc((size_t)32 * 1024 * 64 * 2);

    // Phase A: max-pool (streams 128 MiB of global_f)
    maxpool_kernel<<<2048, 256, 0, stream>>>(global_f, gfeat, 32 * 512, 2048);

    // Prep: BN folds + padded f16 weights
    prep_tok_bn<<<256, 256, 0, stream>>>(tbn_g, tbn_b, tbn_m, tbn_v, s_tok, t_tok, 65536);
    prep_w<<<32, 256, 0, stream>>>(q_w, nullptr, qbn_g, qbn_b, qbn_m, qbn_v, Wq, Bq, 122, 128, 64, 64);
    prep_w<<<32, 256, 0, stream>>>(k_w, nullptr, kbn_g, kbn_b, kbn_m, kbn_v, Wk, Bk, 122, 128, 64, 64);
    prep_w<<<32, 256, 0, stream>>>(v_w, nullptr, vbn_g, vbn_b, vbn_m, vbn_v, Wv, Bv, 122, 128, 64, 64);
    prep_w<<<32, 256, 0, stream>>>(f_w, nullptr, fbn_g, fbn_b, fbn_m, fbn_v, Wf, Bf, 61, 64, 122, 128);
    prep_w<<<64, 256, 0, stream>>>(c1_w, c1_b, nullptr, nullptr, nullptr, nullptr, W1, B1, 256, 256, 61, 64);
    prep_w<<<64, 256, 0, stream>>>(c2_w, c2_b, nullptr, nullptr, nullptr, nullptr, W2, B2, 64, 64, 256, 256);
    prep_w<<<8,  256, 0, stream>>>(c3_w, c3_b, nullptr, nullptr, nullptr, nullptr, W3, B3, 3, 16, 64, 64);

    // Phase B: token GEMM + BN + ReLU (streams 128 MiB of token_w)
    token_kernel<<<1024, 256, 0, stream>>>(token_w, token_b, s_tok, t_tok, gfeat, tokT);

    dim3 blk(256);
    // q/k/v = relu(Wqkv' @ tok + bias): M=128, N=1024, K=64
    wmma_gemm_kernel<<<dim3(16, 32), blk, 0, stream>>>(Wq, 64, 0, tokT, 64, 1024 * 64,
        qT, 128, 1024 * 128, Bq, 8, 64, 64, 128, 1, 1);          // qT[b][n][c]
    wmma_gemm_kernel<<<dim3(16, 32), blk, 0, stream>>>(Wk, 64, 0, tokT, 64, 1024 * 64,
        kC, 1024, 128 * 1024, Bk, 8, 64, 64, 128, 1, 0);         // kC[b][c][m]
    wmma_gemm_kernel<<<dim3(16, 32), blk, 0, stream>>>(Wv, 64, 0, tokT, 64, 1024 * 64,
        vC, 1024, 128 * 1024, Bv, 8, 64, 64, 128, 1, 0);         // vC[b][c][m]

    // Linear-attention trick: KV = k @ v^T (TDM-staged B, TENSORcnt pipelined)
    kv_kernel<<<dim3(2, 32), blk, 0, stream>>>(kC, vC, KVt);     // KVt[c][c']

    // val^T = q^T @ KV (1024x128, K=128)
    wmma_gemm_kernel<<<dim3(16, 32), blk, 0, stream>>>(qT, 128, 1024 * 128, KVt, 128, 128 * 128,
        valT, 128, 1024 * 128, nullptr, 64, 8, 128, 1024, 0, 0); // valT[b][n][c]

    // fused = relu(Wf' @ val + bf): M=64, K=128
    wmma_gemm_kernel<<<dim3(16, 32), blk, 0, stream>>>(Wf, 128, 0, valT, 128, 1024 * 128,
        fusT, 64, 1024 * 64, Bf, 4, 64, 128, 64, 1, 1);          // fusT[b][n][c]

    // head: c1 -> c2 -> c3(sigmoid-0.5, f32 into d_out as (B,1024,3))
    wmma_gemm_kernel<<<dim3(32, 32), blk, 0, stream>>>(W1, 64, 0, fusT, 64, 1024 * 64,
        h1T, 256, 1024 * 256, B1, 16, 64, 64, 256, 1, 1);
    wmma_gemm_kernel<<<dim3(16, 32), blk, 0, stream>>>(W2, 256, 0, h1T, 256, 1024 * 256,
        h2T, 64, 1024 * 64, B2, 4, 64, 256, 64, 1, 1);
    wmma_gemm_kernel<<<dim3(16, 32), blk, 0, stream>>>(W3, 64, 0, h2T, 64, 1024 * 64,
        d_out, 3, 1024 * 3, B3, 1, 64, 64, 3, 2, 3);
}
